// CausalSelfAttention_22703197127087
// MI455X (gfx1250) — compile-verified, matrix operand order checked by reading, not run on hardware
//
#include <hip/hip_runtime.h>

// ---------------------------------------------------------------------------
// CDNA5 (gfx1250, wave32) causal self-attention, bf16 WMMA pipeline.
//   X:(4,2048,1024) f32, W*:(1024,1024) f32 (out = x@W^T + b), out f32.
// Workspace layout (ushort = raw bf16 bits), ~92.3 MB total:
//   [Xbf 8M][Q 8M][K 8M][Vt 8M][Ctx 8M][Wq 1M][Wk 1M][Wv 1M][Wo 1M] elements
//
// GEMM wave tile is 32x64 (2x4 accumulators = 64 VGPRs, 6 frags in flight):
// this stays well under the 256 directly-addressable VGPRs so the backend
// software-pipelines global_load_b128 under v_wmma with no scratch spills
// (a 64x64 tile was tried and spilled -- strictly worse).
// ---------------------------------------------------------------------------

typedef __attribute__((ext_vector_type(16))) __bf16 v16bf;
typedef __attribute__((ext_vector_type(8)))  float  v8f;

struct alignas(16) U4 { unsigned x, y, z, w; };
struct U8 { U4 lo, hi; };
struct alignas(16) F4 { float x, y, z, w; };

static __device__ __forceinline__ unsigned short f2bf(float f) {
  unsigned u = __builtin_bit_cast(unsigned, f);
  return (unsigned short)((u + 0x7FFFu + ((u >> 16) & 1u)) >> 16);
}

// A/B fragment loader, 16x32 bf16 operand, 16 elems per lane:
// elements [p, p+8) and [p+16, p+24)  (caller bakes in row base + k-half).
static __device__ __forceinline__ v16bf ldfrag(const unsigned short* p) {
  U8 f;
  f.lo = *reinterpret_cast<const U4*>(p);
  f.hi = *reinterpret_cast<const U4*>(p + 16);
  return __builtin_bit_cast(v16bf, f);
}

static __device__ __forceinline__ v8f wmma_bf16(v16bf a, v16bf b, v8f c) {
  return __builtin_amdgcn_wmma_f32_16x16x32_bf16(
      /*neg_a=*/false, a, /*neg_b=*/false, b,
      /*c_mod=*/(short)0, c, /*reuse_a=*/false, /*reuse_b=*/false);
}

// ---------------------------------------------------------------------------
// fp32 -> bf16 bits conversion, 8 elems/thread/iter (b128 in, b128 out)
// ---------------------------------------------------------------------------
__global__ void cvt_k(const float* __restrict__ in, unsigned short* __restrict__ out, int n) {
  const int stride = gridDim.x * blockDim.x * 8;
  for (int i = (blockIdx.x * blockDim.x + threadIdx.x) * 8; i < n; i += stride) {
    F4 f0 = *reinterpret_cast<const F4*>(in + i);
    F4 f1 = *reinterpret_cast<const F4*>(in + i + 4);
    U4 o;
    o.x = (unsigned)f2bf(f0.x) | ((unsigned)f2bf(f0.y) << 16);
    o.y = (unsigned)f2bf(f0.z) | ((unsigned)f2bf(f0.w) << 16);
    o.z = (unsigned)f2bf(f1.x) | ((unsigned)f2bf(f1.y) << 16);
    o.w = (unsigned)f2bf(f1.z) | ((unsigned)f2bf(f1.w) << 16);
    *reinterpret_cast<U4*>(out + i) = o;
  }
}

// ---------------------------------------------------------------------------
// GEMM: out[m,n] = sum_d A[m,d] * W[n,d] + bias[n]     (M=8192, N=K=1024)
// Wave tile 32x64 (2x4 accumulators), block = 8 waves -> 128x128 tile.
// outb!=null : bf16 output; vtrans=1 stores per-batch transposed (for V).
// outf!=null : fp32 output (final projection into d_out).
// ---------------------------------------------------------------------------
__global__ void gemm_k(const unsigned short* __restrict__ A,
                       const unsigned short* __restrict__ W,
                       const float* __restrict__ bias,
                       unsigned short* __restrict__ outb,
                       float* __restrict__ outf, int vtrans) {
  const int lane = threadIdx.x & 31;
  const int wave = threadIdx.x >> 5;
  const int m0 = blockIdx.x * 128 + (wave >> 1) * 32;
  const int n0 = blockIdx.y * 128 + (wave & 1) * 64;
  const int rsel = lane & 15;           // A: row-in-tile, B: col-in-tile, C: col
  const int kh   = (lane >> 4) * 8;     // k-half selector
  const int rb   = (lane >> 4) * 8;     // C-layout row base

  v8f acc[2][4];
#pragma unroll
  for (int mi = 0; mi < 2; ++mi)
#pragma unroll
    for (int j = 0; j < 4; ++j) {
      float bv = bias[n0 + j * 16 + rsel];
#pragma unroll
      for (int r = 0; r < 8; ++r) acc[mi][j][r] = bv;
    }

  const unsigned short* Ar0 = A + (size_t)(m0 + rsel) * 1024;
  const unsigned short* Ar1 = A + (size_t)(m0 + 16 + rsel) * 1024;
  const unsigned short* Wr0 = W + (size_t)(n0 + 0 * 16 + rsel) * 1024;
  const unsigned short* Wr1 = W + (size_t)(n0 + 1 * 16 + rsel) * 1024;
  const unsigned short* Wr2 = W + (size_t)(n0 + 2 * 16 + rsel) * 1024;
  const unsigned short* Wr3 = W + (size_t)(n0 + 3 * 16 + rsel) * 1024;

  for (int k = 0; k < 1024; k += 32) {
    v16bf a0 = ldfrag(Ar0 + k + kh);
    v16bf a1 = ldfrag(Ar1 + k + kh);
    v16bf b0 = ldfrag(Wr0 + k + kh);
    v16bf b1 = ldfrag(Wr1 + k + kh);
    v16bf b2 = ldfrag(Wr2 + k + kh);
    v16bf b3 = ldfrag(Wr3 + k + kh);
    acc[0][0] = wmma_bf16(a0, b0, acc[0][0]);
    acc[1][0] = wmma_bf16(a1, b0, acc[1][0]);
    acc[0][1] = wmma_bf16(a0, b1, acc[0][1]);
    acc[1][1] = wmma_bf16(a1, b1, acc[1][1]);
    acc[0][2] = wmma_bf16(a0, b2, acc[0][2]);
    acc[1][2] = wmma_bf16(a1, b2, acc[1][2]);
    acc[0][3] = wmma_bf16(a0, b3, acc[0][3]);
    acc[1][3] = wmma_bf16(a1, b3, acc[1][3]);
  }

#pragma unroll
  for (int mi = 0; mi < 2; ++mi)
#pragma unroll
    for (int j = 0; j < 4; ++j) {
      const int col = n0 + j * 16 + rsel;
      const int rowbase = m0 + mi * 16 + rb;
#pragma unroll
      for (int r = 0; r < 8; ++r) {
        const int row = rowbase + r;
        const float v = acc[mi][j][r];
        if (outf) {
          outf[(size_t)row * 1024 + col] = v;
        } else if (vtrans) {           // V^T per batch: Vt[b][col][row%2048]
          const int bb = row >> 11, rr = row & 2047;
          outb[(size_t)bb * 1024 * 2048 + (size_t)col * 2048 + rr] = f2bf(v);
        } else {
          outb[(size_t)row * 1024 + col] = f2bf(v);
        }
      }
    }
}

// ---------------------------------------------------------------------------
// Flash-style causal attention. One block (8 waves) per 16 query rows.
//  - waves split d=1024 into 8 chunks for S = Q K^T, reduce via ds_add_f32
//  - wave 0 does online softmax (scale 1/32; ref's +1.0 on unmasked entries is
//    a per-row constant -> softmax-invariant, dropped), writes P(bf16) + alpha
//  - each wave owns a 128-col slice of ctx, updates with P x V^T WMMAs
// ---------------------------------------------------------------------------
__global__ void attn_k(const unsigned short* __restrict__ Q,
                       const unsigned short* __restrict__ K,
                       const unsigned short* __restrict__ Vt,
                       unsigned short* __restrict__ Ctx) {
  __shared__ alignas(16) float          sbuf[16 * 32];   // S reduction (f32)
  __shared__ alignas(16) unsigned short pbuf[16 * 32];   // P tile (bf16 bits)
  __shared__ alignas(16) float          salpha[16];
  __shared__ alignas(16) float          slinv[16];

  const int tid  = threadIdx.x;
  const int lane = tid & 31;
  const int wave = tid >> 5;
  const int b    = blockIdx.x >> 7;
  const int q0   = (blockIdx.x & 127) * 16;
  const int rowa = lane & 15;
  const int kh   = (lane >> 4) * 8;
  const int rb   = (lane >> 4) * 8;
  const int f0   = wave * 128;     // ctx feature slice owned by this wave
  const int db   = wave * 128;     // d-chunk of the score contraction

  const unsigned short* Qrow  = Q  + ((size_t)b * 2048 + q0 + rowa) * 1024;
  const unsigned short* Kbase = K  + (size_t)b * 2048 * 1024;
  const unsigned short* Vbase = Vt + (size_t)b * 1024 * 2048;

  v8f cacc[8];
#pragma unroll
  for (int nt = 0; nt < 8; ++nt)
#pragma unroll
    for (int r = 0; r < 8; ++r) cacc[nt][r] = 0.f;

  // Q fragments are invariant over key steps: preload 4 (d-chunk of 128).
  v16bf aq[4];
#pragma unroll
  for (int d = 0; d < 4; ++d) aq[d] = ldfrag(Qrow + db + d * 32 + kh);

  float mrow = -1e30f, lrow = 0.f;   // wave0 lanes 0..15 softmax state

  const int nsteps = (q0 + 16 + 31) >> 5;   // keys processed 32 at a time
  for (int s = 0; s < nsteps; ++s) {
    const int kt = s * 32;

    __syncthreads();                 // prior P consumed, sbuf reusable
    sbuf[tid]       = 0.f;
    sbuf[tid + 256] = 0.f;
    __syncthreads();

    // Pull next key-step's K/V lines toward the WGP (global_prefetch_b8).
    if (s + 1 < nsteps) {
      __builtin_prefetch(Kbase + (size_t)(kt + 32 + rowa) * 1024 + db + kh, 0, 0);
      __builtin_prefetch(Vbase + (size_t)(f0 + rowa) * 2048 + kt + 32 + kh, 0, 0);
    }

    // Partial S(16x32) over this wave's d-chunk, accumulate into LDS.
#pragma unroll
    for (int c = 0; c < 2; ++c) {
      v8f sacc;
#pragma unroll
      for (int r = 0; r < 8; ++r) sacc[r] = 0.f;
#pragma unroll
      for (int d = 0; d < 4; ++d) {
        v16bf bk = ldfrag(Kbase + (size_t)(kt + c * 16 + rowa) * 1024 + db + d * 32 + kh);
        sacc = wmma_bf16(aq[d], bk, sacc);
      }
#pragma unroll
      for (int r = 0; r < 8; ++r)
        atomicAdd(&sbuf[(rb + r) * 32 + c * 16 + rowa], sacc[r]);
    }
    __syncthreads();

    // Online softmax on the completed 16x32 S tile (wave 0, one row/lane).
    if (wave == 0 && lane < 16) {
      const int row = lane, qabs = q0 + row;
      float rmax = -1e30f;
      for (int j = 0; j < 32; ++j) {
        float sv = sbuf[row * 32 + j] * 0.03125f;      // 1/sqrt(1024)
        if (kt + j > qabs) sv = -1e30f;                // causal mask
        rmax = fmaxf(rmax, sv);
      }
      const float mnew = fmaxf(mrow, rmax);
      const float al   = __expf(mrow - mnew);
      float psum = 0.f;
      for (int j = 0; j < 32; ++j) {
        float sv = sbuf[row * 32 + j] * 0.03125f;
        float p  = (kt + j > qabs) ? 0.f : __expf(sv - mnew);
        psum += p;
        pbuf[row * 32 + j] = f2bf(p);
      }
      lrow = lrow * al + psum;
      mrow = mnew;
      salpha[row] = al;
    }
    __syncthreads();

    // Rescale running ctx and accumulate P(16x32) x V^T-slice(32x128).
    F4 a0 = *reinterpret_cast<const F4*>(&salpha[rb]);
    F4 a1 = *reinterpret_cast<const F4*>(&salpha[rb + 4]);
    float af[8] = {a0.x, a0.y, a0.z, a0.w, a1.x, a1.y, a1.z, a1.w};
    v16bf pf = ldfrag(pbuf + rowa * 32 + kh);          // ds_load_b128 x2
#pragma unroll
    for (int nt = 0; nt < 8; ++nt) {
#pragma unroll
      for (int r = 0; r < 8; ++r) cacc[nt][r] *= af[r];
      v16bf bv = ldfrag(Vbase + (size_t)(f0 + nt * 16 + rowa) * 2048 + kt + kh);
      cacc[nt] = wmma_bf16(pf, bv, cacc[nt]);
    }
  }

  if (wave == 0 && lane < 16) slinv[lane] = 1.0f / lrow;
  __syncthreads();

  F4 l0 = *reinterpret_cast<const F4*>(&slinv[rb]);
  F4 l1 = *reinterpret_cast<const F4*>(&slinv[rb + 4]);
  float lf[8] = {l0.x, l0.y, l0.z, l0.w, l1.x, l1.y, l1.z, l1.w};
#pragma unroll
  for (int nt = 0; nt < 8; ++nt) {
    const int col = f0 + nt * 16 + rowa;
#pragma unroll
    for (int r = 0; r < 8; ++r) {
      const int row = q0 + rb + r;
      Ctx[((size_t)b * 2048 + row) * 1024 + col] = f2bf(cacc[nt][r] * lf[r]);
    }
  }
}

// ---------------------------------------------------------------------------
extern "C" void kernel_launch(void* const* d_in, const int* in_sizes, int n_in,
                              void* d_out, int out_size, void* d_ws, size_t ws_size,
                              hipStream_t stream) {
  (void)in_sizes; (void)n_in; (void)out_size; (void)ws_size;

  const float* X  = (const float*)d_in[0];
  const float* Wq = (const float*)d_in[1];
  const float* bq = (const float*)d_in[2];
  const float* Wk = (const float*)d_in[3];
  const float* bk = (const float*)d_in[4];
  const float* Wv = (const float*)d_in[5];
  const float* bv = (const float*)d_in[6];
  const float* Wo = (const float*)d_in[7];
  const float* bo = (const float*)d_in[8];
  float* out = (float*)d_out;

  const size_t SZ_X = 8192ull * 1024;   // 8,388,608 elements
  const size_t SZ_W = 1024ull * 1024;
  unsigned short* ws  = (unsigned short*)d_ws;   // needs ~92.3 MB
  unsigned short* Xbf = ws;
  unsigned short* Qb  = ws + 1 * SZ_X;
  unsigned short* Kb  = ws + 2 * SZ_X;
  unsigned short* Vtb = ws + 3 * SZ_X;
  unsigned short* Ctx = ws + 4 * SZ_X;
  unsigned short* Wqb = ws + 5 * SZ_X;
  unsigned short* Wkb = Wqb + SZ_W;
  unsigned short* Wvb = Wkb + SZ_W;
  unsigned short* Wob = Wvb + SZ_W;

  cvt_k<<<2048, 256, 0, stream>>>(X,  Xbf, (int)SZ_X);
  cvt_k<<<512,  256, 0, stream>>>(Wq, Wqb, (int)SZ_W);
  cvt_k<<<512,  256, 0, stream>>>(Wk, Wkb, (int)SZ_W);
  cvt_k<<<512,  256, 0, stream>>>(Wv, Wvb, (int)SZ_W);
  cvt_k<<<512,  256, 0, stream>>>(Wo, Wob, (int)SZ_W);

  dim3 gg(64, 8);   // 128x128 block tiles over 8192x1024
  gemm_k<<<gg, 256, 0, stream>>>(Xbf, Wqb, bq, Qb,  nullptr, 0);
  gemm_k<<<gg, 256, 0, stream>>>(Xbf, Wkb, bk, Kb,  nullptr, 0);
  gemm_k<<<gg, 256, 0, stream>>>(Xbf, Wvb, bv, Vtb, nullptr, 1);   // V stored transposed

  attn_k<<<512, 256, 0, stream>>>(Qb, Kb, Vtb, Ctx);

  gemm_k<<<gg, 256, 0, stream>>>(Ctx, Wob, bo, nullptr, out, 0);
}
